// MultiScaleSlidingWindow_23175643529563
// MI455X (gfx1250) — matrix operand order repaired and checked by reading, not compiled, not run on hardware
//
#include <hip/hip_runtime.h>
#include <hip/hip_bf16.h>
#include <math.h>

// ---------------------------------------------------------------------------
// MultiScaleSlidingWindow fused kernel for MI455X (gfx1250, wave32, WMMA)
//
//   For each scale i in {4,8,16}:
//     y  = x @ W[i]^T + b[i]         (65536 x 1024 x 1024 GEMM)
//     y  = LayerNorm_D(y) * gamma + beta ; gelu_exact(y)
//     out[i][b][d] = sum_n y[b,n,d] * w_i[n]
//
// GEMM on v_wmma_f32_16x16x32_bf16 with 3-term bf16 split (xh*Wh + xh*Wl +
// xl*Wh) for ~fp32 accuracy.  Fully fused: y never touches HBM.
// M-tile = 64 tokens/WG: each pre-packed B fragment feeds 4 row blocks
// (48 WMMA : 32 b128-loads per K-step).  Epilogue parks y in LDS (reusing
// the dead x-staging space) so the 128 accumulator VGPRs retire before the
// LayerNorm/GELU passes -> no register-pressure spills.
// ---------------------------------------------------------------------------

typedef __attribute__((ext_vector_type(16))) __bf16 v16bf;
typedef __attribute__((ext_vector_type(8)))  float  v8f;

#define D_DIM   1024
#define N_TOK   2048
#define B_SZ    32
#define LDSW    1032            // padded bf16 row stride (16B-aligned rows)
#define YW      1028            // padded fp32 row stride for y tile
#define M_TILE  64
#define RB_CT   4               // row blocks of 16 tokens
#define THREADS 512             // 16 waves; wave owns 4 N-tiles (64 cols)
#define SH_BYTES (2 * M_TILE * LDSW * 2)   // 264192 B (y phase needs less)

__device__ __forceinline__ void split_bf16(float x, __bf16& h, __bf16& l) {
  h = (__bf16)x;                       // RNE f32 -> bf16
  l = (__bf16)(x - (float)h);          // residual
}

// ---------------------------------------------------------------------------
// Pre-pack W[i][e][d] (fp32) into WMMA B-fragment layout, bf16 hi/lo.
// B fragment (K=32 x N=16, bf16): lane = N + 16*h, element e -> K = 8h+e (e<8), 16+8h+(e-8) (e>=8).
// Fragment index: (((scale*32 + ktile)*64 + ntile)*32 + lane)*16 + e
// ---------------------------------------------------------------------------
__global__ __launch_bounds__(256) void pack_w_kernel(const float* __restrict__ W,
                                                     __bf16* __restrict__ whi,
                                                     __bf16* __restrict__ wlo) {
  int gid = blockIdx.x * 256 + threadIdx.x;     // 0 .. 3*32*64*32-1
  int lane = gid & 31;
  int nt   = (gid >> 5)  & 63;
  int kk   = (gid >> 11) & 31;
  int i    =  gid >> 16;

  int ncol  = nt * 16 + (lane & 15);
  int kbase = kk * 32 + ((lane >= 16) ? 8 : 0);
  const float* src = W + ((size_t)i * D_DIM + ncol) * D_DIM + kbase;
  size_t off = (size_t)gid * 16;
#pragma unroll
  for (int e = 0; e < 16; ++e) {
    __bf16 h, l;
    split_bf16(src[e + (e & 8)], h, l);
    whi[off + e] = h;
    wlo[off + e] = l;
  }
}

__global__ void zero_out_kernel(float* __restrict__ out, int n) {
  int i = blockIdx.x * 256 + threadIdx.x;
  if (i < n) out[i] = 0.0f;
}

// ---------------------------------------------------------------------------
// Main fused kernel.
//   grid = (1024 tiles of 64 tokens, 3 scales), block = 512 (16 waves)
//   wave w owns N-tiles  w*4 .. w*4+3  (64 output columns)
// ---------------------------------------------------------------------------
__global__ __launch_bounds__(THREADS) void msw_fused_kernel(
    const float* __restrict__ x,
    const float* __restrict__ bias,
    const float* __restrict__ gamma,
    const float* __restrict__ beta,
    const __bf16* __restrict__ whi,
    const __bf16* __restrict__ wlo,
    float* __restrict__ out) {
  extern __shared__ char smem[];
  __bf16* shi = (__bf16*)smem;                  // [64][LDSW] bf16 hi of x tile
  __bf16* slo = shi + M_TILE * LDSW;            // [64][LDSW] bf16 lo of x tile
  float*  yb  = (float*)smem;                   // [64][YW]  y tile (after GEMM)
  float*  red = yb + M_TILE * YW;               // [128] LN sum(64) / sumsq(64)

  const int tid  = threadIdx.x;
  const int lane = tid & 31;
  const int wave = tid >> 5;
  const int tile = blockIdx.x;             // 0..1023
  const int isc  = blockIdx.y;             // scale index
  const int s    = 4 << isc;               // 4, 8, 16
  const int bidx = tile >> 5;              // batch row (32 tiles per row)
  const int t0   = (tile & 31) * M_TILE;   // first token of tile

  // ---- stage x tile (64 tokens x 1024) into LDS as bf16 hi/lo -------------
  const float* xrow = x + ((size_t)bidx * N_TOK + t0) * D_DIM;
#pragma unroll
  for (int it = 0; it < 32; ++it) {
    int idx = it * THREADS + tid;          // float4 index, 256 per row
    int r   = idx >> 8;
    int c4  = idx & 255;
    float4 v = ((const float4*)(xrow + (size_t)r * D_DIM))[c4];
    union { __bf16 h[4]; uint2 u; } ph, pl;
    split_bf16(v.x, ph.h[0], pl.h[0]);
    split_bf16(v.y, ph.h[1], pl.h[1]);
    split_bf16(v.z, ph.h[2], pl.h[2]);
    split_bf16(v.w, ph.h[3], pl.h[3]);
    *(uint2*)(shi + r * LDSW + c4 * 4) = ph.u;
    *(uint2*)(slo + r * LDSW + c4 * 4) = pl.u;
  }
  __syncthreads();

  // ---- GEMM: K loop of 32-wide bf16 WMMA steps ----------------------------
  v8f acc[RB_CT][4];
#pragma unroll
  for (int rb = 0; rb < RB_CT; ++rb)
#pragma unroll
    for (int j = 0; j < 4; ++j) acc[rb][j] = (v8f)0.0f;

  const int halfoff = (lane >= 16) ? 8 : 0;   // A: K offset; C/D: row offset
  const int m = lane & 15;                    // A row / B column within tile

  union AFrag { uint4 q[2]; v16bf v; };

  for (int kk = 0; kk < 32; ++kk) {
    AFrag ahi[RB_CT], alo[RB_CT];
#pragma unroll
    for (int rb = 0; rb < RB_CT; ++rb) {
      const __bf16* ph = shi + (rb * 16 + m) * LDSW + kk * 32 + halfoff;
      const __bf16* pl = slo + (rb * 16 + m) * LDSW + kk * 32 + halfoff;
      ahi[rb].q[0] = *(const uint4*)(ph);        // K 0..7   (ds_load_b128)
      ahi[rb].q[1] = *(const uint4*)(ph + 16);   // K 16..23
      alo[rb].q[0] = *(const uint4*)(pl);
      alo[rb].q[1] = *(const uint4*)(pl + 16);
    }

#pragma unroll
    for (int j = 0; j < 4; ++j) {
      int nt = wave * 4 + j;
      size_t off = ((((size_t)isc * 32 + kk) * 64 + nt) * 32 + lane) * 16;
      v16bf bhi = *(const v16bf*)(whi + off);    // 2x global_load_b128
      v16bf blo = *(const v16bf*)(wlo + off);
#pragma unroll
      for (int rb = 0; rb < RB_CT; ++rb) {
        acc[rb][j] = __builtin_amdgcn_wmma_f32_16x16x32_bf16(
            false, ahi[rb].v, false, bhi, (short)0, acc[rb][j], false, false);
        acc[rb][j] = __builtin_amdgcn_wmma_f32_16x16x32_bf16(
            false, ahi[rb].v, false, blo, (short)0, acc[rb][j], false, false);
        acc[rb][j] = __builtin_amdgcn_wmma_f32_16x16x32_bf16(
            false, alo[rb].v, false, bhi, (short)0, acc[rb][j], false, false);
      }
    }
  }
  __syncthreads();   // x tile dead; smem becomes the y tile

  // ---- bias add; park y in LDS (retires all accumulator VGPRs) ------------
#pragma unroll
  for (int j = 0; j < 4; ++j) {
    int col = (wave * 4 + j) * 16 + m;
    float bv = bias[isc * D_DIM + col];
#pragma unroll
    for (int rb = 0; rb < RB_CT; ++rb)
#pragma unroll
      for (int r = 0; r < 8; ++r) {
        int row = rb * 16 + halfoff + r;
        yb[row * YW + col] = acc[rb][j][r] + bv;
      }
  }
  if (tid < 128) red[tid] = 0.0f;
  __syncthreads();

  // ---- LN statistics: 8 threads per row, contiguous b128 reads ------------
  {
    int row = tid >> 3;                    // 0..63
    int seg = tid & 7;                     // 128 columns each
    const float* p = yb + row * YW + seg * 128;
    float sm = 0.0f, sq = 0.0f;
#pragma unroll
    for (int q = 0; q < 32; ++q) {         // 32 x float4
      float4 v = ((const float4*)p)[q];
      sm += v.x + v.y + v.z + v.w;
      sq += v.x * v.x + v.y * v.y + v.z * v.z + v.w * v.w;
    }
    atomicAdd(&red[row], sm);              // ds_add_f32, 8 adds per row
    atomicAdd(&red[64 + row], sq);
  }
  __syncthreads();

  if (tid < 64) {                          // precompute mu / rsigma in place
    float mn  = red[tid] * (1.0f / (float)D_DIM);
    float var = red[64 + tid] * (1.0f / (float)D_DIM) - mn * mn;
    red[tid]      = mn;
    red[64 + tid] = rsqrtf(var + 1e-5f);
  }
  __syncthreads();

  // ---- LN apply + gamma/beta + exact GELU + windowed token reduction ------
  {
    const int nw = N_TOK - s + 1;
    int col0 = tid;                        // thread owns cols tid, tid+512
    int col1 = tid + 512;
    float g0 = gamma[isc * D_DIM + col0], b0 = beta[isc * D_DIM + col0];
    float g1 = gamma[isc * D_DIM + col1], b1 = beta[isc * D_DIM + col1];
    float o0 = 0.0f, o1 = 0.0f;
    for (int r = 0; r < M_TILE; ++r) {
      float mu = red[r];                   // LDS broadcast reads
      float rs = red[64 + r];
      int tok = t0 + r;
      int c = min(min(tok + 1, s), min(nw, N_TOK - tok));
      float w = (float)c / ((float)nw * (float)s);
      float v0 = (yb[r * YW + col0] - mu) * rs * g0 + b0;
      float v1 = (yb[r * YW + col1] - mu) * rs * g1 + b1;
      o0 += 0.5f * v0 * (1.0f + erff(v0 * 0.70710678118654752f)) * w;
      o1 += 0.5f * v1 * (1.0f + erff(v1 * 0.70710678118654752f)) * w;
    }
    float* obase = out + ((size_t)isc * B_SZ + bidx) * D_DIM;
    atomicAdd(&obase[col0], o0);
    atomicAdd(&obase[col1], o1);
  }
}

// ---------------------------------------------------------------------------
extern "C" void kernel_launch(void* const* d_in, const int* in_sizes, int n_in,
                              void* d_out, int out_size, void* d_ws, size_t ws_size,
                              hipStream_t stream) {
  const float* x     = (const float*)d_in[0];  // [32,2048,1024]
  const float* W     = (const float*)d_in[1];  // [3,1024,1024]
  const float* bias  = (const float*)d_in[2];  // [3,1024]
  const float* gamma = (const float*)d_in[3];  // [3,1024]
  const float* beta  = (const float*)d_in[4];  // [3,1024]
  float* out = (float*)d_out;                  // [3,32,1024] concat

  __bf16* whi = (__bf16*)d_ws;                        // 6 MB
  __bf16* wlo = whi + (size_t)3 * D_DIM * D_DIM;      // 6 MB

  // allow > default dynamic LDS (264.2 KB of the 320 KB WGP budget)
  static int attr_set = 0;
  if (!attr_set) {
    (void)hipFuncSetAttribute((const void*)msw_fused_kernel,
                              hipFuncAttributeMaxDynamicSharedMemorySize,
                              SH_BYTES);
    attr_set = 1;
  }

  // 1) pack W into WMMA B-fragment layout (bf16 hi/lo)
  pack_w_kernel<<<(3 * 32 * 64 * 32) / 256, 256, 0, stream>>>(W, whi, wlo);

  // 2) zero the (poisoned) output for the atomic reduction
  zero_out_kernel<<<(out_size + 255) / 256, 256, 0, stream>>>(out, out_size);

  // 3) fused GEMM + LN + GELU + windowed token reduction
  dim3 grid((B_SZ * N_TOK) / M_TILE, 3);
  msw_fused_kernel<<<grid, THREADS, SH_BYTES, stream>>>(x, bias, gamma, beta,
                                                        whi, wlo, out);
}
